// MeshRefineNet_6889127543462
// MI455X (gfx1250) — matrix operand order repaired and checked by reading, not compiled, not run on hardware
//
#include <hip/hip_runtime.h>

// ---------------------------------------------------------------------------
// Types for CDNA5 WMMA (wave32): v_wmma_f32_16x16x32_bf16
// ---------------------------------------------------------------------------
typedef __attribute__((ext_vector_type(16))) __bf16 bf16x16;
typedef __attribute__((ext_vector_type(8)))  __bf16 bf16x8;
typedef __attribute__((ext_vector_type(8)))  float  f32x8;
typedef __attribute__((ext_vector_type(16))) float  f32x16;

__device__ __forceinline__ __bf16 f2bf(float f) {
  unsigned u = __builtin_bit_cast(unsigned, f);
  unsigned r = (u + 0x7FFFu + ((u >> 16) & 1u)) >> 16;   // round-to-nearest-even
  return __builtin_bit_cast(__bf16, (unsigned short)r);
}

#define F 128            // feature width / K depth
#define NSQ (128 * 128)  // elements per 128x128 weight matrix
#define SREC 24          // padded LDS record stride (elements) per lane (48B)

// ---------------------------------------------------------------------------
// One-shot weight transpose + f32->bf16 convert.
// Layout: Wt[col*128 + k]  (column-major over K) so a lane's B fragment
// (16 consecutive K values) is one contiguous 32B load.
// wt blocks: [W0_0, W1_0, W0_1, W1_1, W0_2, W1_2] (6*16384) then
//            [W0_3 padded 16x128, W1_3 padded 16x128] (2*2048)
// ---------------------------------------------------------------------------
__global__ __launch_bounds__(256)
void prep_weights(const float* __restrict__ W00, const float* __restrict__ W10,
                  const float* __restrict__ W01, const float* __restrict__ W11,
                  const float* __restrict__ W02, const float* __restrict__ W12,
                  const float* __restrict__ W03, const float* __restrict__ W13,
                  __bf16* __restrict__ wt) {
  int idx = blockIdx.x * 256 + threadIdx.x;
  if (idx < 6 * NSQ) {
    int m = idx / NSQ;
    int r = idx % NSQ;
    int n = r >> 7, k = r & 127;
    const float* W = (m == 0) ? W00 : (m == 1) ? W10 : (m == 2) ? W01
                   : (m == 3) ? W11 : (m == 4) ? W02 : W12;
    wt[m * NSQ + n * F + k] = f2bf(W[k * F + n]);
  } else if (idx < 6 * NSQ + 2 * 16 * F) {
    int r  = idx - 6 * NSQ;
    int m  = r / (16 * F);
    int rr = r % (16 * F);
    int n = rr >> 7, k = rr & 127;
    const float* W = m ? W13 : W03;
    float v = (n < 3) ? W[k * 3 + n] : 0.0f;
    wt[6 * NSQ + m * (16 * F) + n * F + k] = f2bf(v);
  }
}

// ---------------------------------------------------------------------------
// Dual GEMM: H0 = act(A) @ W0 + b0 ; H1 = act(A) @ W1 + b1   (dout = 128)
// One wave owns a 16x16 tile, 8 waves cover 16 rows x 128 cols.
// The 16x128 A tile (identical for all 8 waves) is converted to bf16 ONCE
// per block into LDS in WMMA fragment order, then each lane reads its
// fragment with two ds_load_b128.
// ---------------------------------------------------------------------------
__global__ __launch_bounds__(256)
void gemm_dual128(const float* __restrict__ A,
                  const __bf16* __restrict__ Wt0, const float* __restrict__ bias0,
                  const __bf16* __restrict__ Wt1, const float* __restrict__ bias1,
                  float* __restrict__ H0, float* __restrict__ H1, int relu) {
  // sA[kstep][lane record]: record = 16 bf16 (fragment for that lane), padded
  // to SREC elements (48B) so half-wave ds reads hit distinct banks.
  __shared__ __bf16 sA[4 * 32 * SREC];

  const int tid  = threadIdx.x;
  const int row0 = blockIdx.x * 16;

  // ---- Cooperative A-tile convert: one thread = one 8-float run ----
  {
    int kstep = tid >> 6;          // 0..3
    int rem   = tid & 63;
    int rlane = rem >> 1;          // fragment-owner lane 0..31
    int half  = rem & 1;           // low/high 8 elements of the fragment
    int rlm   = rlane & 15;
    int rg    = rlane >> 4;
    // A fragment order used here (k contiguous, same as B): lane group g holds K run
    // [kk+16g, kk+16g+16).
    int kbase = kstep * 32 + 16 * rg + (half ? 8 : 0);
    const float4* p = (const float4*)(A + (size_t)(row0 + rlm) * F + kbase);
    float4 x0 = p[0], x1 = p[1];
    if (relu) {
      x0.x = fmaxf(x0.x, 0.f); x0.y = fmaxf(x0.y, 0.f);
      x0.z = fmaxf(x0.z, 0.f); x0.w = fmaxf(x0.w, 0.f);
      x1.x = fmaxf(x1.x, 0.f); x1.y = fmaxf(x1.y, 0.f);
      x1.z = fmaxf(x1.z, 0.f); x1.w = fmaxf(x1.w, 0.f);
    }
    f32x8 v;
    v[0] = x0.x; v[1] = x0.y; v[2] = x0.z; v[3] = x0.w;
    v[4] = x1.x; v[5] = x1.y; v[6] = x1.z; v[7] = x1.w;
    bf16x8 b = __builtin_convertvector(v, bf16x8);
    *(bf16x8*)&sA[kstep * (32 * SREC) + rlane * SREC + half * 8] = b;
  }
  __syncthreads();

  const int lane = tid & 31;
  const int wave = tid >> 5;
  const int g    = lane >> 4;
  const int lm   = lane & 15;
  const int col  = wave * 16 + lm;   // B/C/D column owned by this lane

  // Bias folded into accumulator init (C layout: N = lane&15 for every VGPR).
  f32x8 acc0, acc1;
  const float bz0 = bias0[col];
  const float bz1 = bias1[col];
#pragma unroll
  for (int r = 0; r < 8; ++r) { acc0[r] = bz0; acc1[r] = bz1; }

#pragma unroll
  for (int kstep = 0; kstep < 4; ++kstep) {
    const int kk = kstep * 32;
    // A fragment from LDS (two 16B reads, joined register-to-register).
    const __bf16* rec = &sA[kstep * (32 * SREC) + lane * SREC];
    bf16x8 alo = *(const bf16x8*)(rec);
    bf16x8 ahi = *(const bf16x8*)(rec + 8);
    bf16x16 a = __builtin_shufflevector(alo, ahi,
                                        0, 1, 2, 3, 4, 5, 6, 7,
                                        8, 9, 10, 11, 12, 13, 14, 15);
    // B fragments: lane group g holds K=[kk+16g, kk+16g+16), column `col`.
    bf16x16 bA = *(const bf16x16*)(Wt0 + (size_t)col * F + kk + 16 * g);
    bf16x16 bB = *(const bf16x16*)(Wt1 + (size_t)col * F + kk + 16 * g);

    acc0 = __builtin_amdgcn_wmma_f32_16x16x32_bf16(false, a, false, bA, (short)0, acc0, false, false);
    acc1 = __builtin_amdgcn_wmma_f32_16x16x32_bf16(false, a, false, bB, (short)0, acc1, false, false);
  }

  // D layout: VGPR r -> (M = r + 8*g, N = lane&15)
#pragma unroll
  for (int r = 0; r < 8; ++r) {
    int m = row0 + r + 8 * g;
    H0[(size_t)m * F + col] = acc0[r];
    H1[(size_t)m * F + col] = acc1[r];
  }
}

// ---------------------------------------------------------------------------
// Layer-3 dual GEMM (dout = 3, padded to 16 cols). A = relu(h) + features.
// Each wave takes a different 16-row slab (no A sharing between waves).
// ---------------------------------------------------------------------------
__global__ __launch_bounds__(256)
void gemm_dual3(const float* __restrict__ A, const float* __restrict__ resid,
                const __bf16* __restrict__ Wt0, const float* __restrict__ bias0,
                const __bf16* __restrict__ Wt1, const float* __restrict__ bias1,
                float* __restrict__ H0, float* __restrict__ H1) {
  const int lane = threadIdx.x & 31;
  const int wave = threadIdx.x >> 5;
  const int g    = lane >> 4;
  const int lm   = lane & 15;
  const int row0 = blockIdx.x * 128 + wave * 16;
  const int row  = row0 + lm;

  f32x8 acc0, acc1;
  const float bz0 = (lm < 3) ? bias0[lm] : 0.0f;
  const float bz1 = (lm < 3) ? bias1[lm] : 0.0f;
#pragma unroll
  for (int r = 0; r < 8; ++r) { acc0[r] = bz0; acc1[r] = bz1; }

  const float* arow = A + (size_t)row * F;
  const float* rrow = resid + (size_t)row * F;

#pragma unroll
  for (int kk = 0; kk < F; kk += 32) {
    const float4* p = (const float4*)(arow + kk + 16 * g);
    const float4* q = (const float4*)(rrow + kk + 16 * g);
    float4 x0 = p[0], x1 = p[1], x2 = p[2], x3 = p[3];
    float4 r0 = q[0], r1 = q[1], r2 = q[2], r3 = q[3];
    f32x16 av;
    av[0]  = fmaxf(x0.x, 0.f) + r0.x; av[1]  = fmaxf(x0.y, 0.f) + r0.y;
    av[2]  = fmaxf(x0.z, 0.f) + r0.z; av[3]  = fmaxf(x0.w, 0.f) + r0.w;
    av[4]  = fmaxf(x1.x, 0.f) + r1.x; av[5]  = fmaxf(x1.y, 0.f) + r1.y;
    av[6]  = fmaxf(x1.z, 0.f) + r1.z; av[7]  = fmaxf(x1.w, 0.f) + r1.w;
    av[8]  = fmaxf(x2.x, 0.f) + r2.x; av[9]  = fmaxf(x2.y, 0.f) + r2.y;
    av[10] = fmaxf(x2.z, 0.f) + r2.z; av[11] = fmaxf(x2.w, 0.f) + r2.w;
    av[12] = fmaxf(x3.x, 0.f) + r3.x; av[13] = fmaxf(x3.y, 0.f) + r3.y;
    av[14] = fmaxf(x3.z, 0.f) + r3.z; av[15] = fmaxf(x3.w, 0.f) + r3.w;
    bf16x16 a = __builtin_convertvector(av, bf16x16);

    bf16x16 bA = *(const bf16x16*)(Wt0 + (size_t)lm * F + kk + 16 * g);
    bf16x16 bB = *(const bf16x16*)(Wt1 + (size_t)lm * F + kk + 16 * g);

    acc0 = __builtin_amdgcn_wmma_f32_16x16x32_bf16(false, a, false, bA, (short)0, acc0, false, false);
    acc1 = __builtin_amdgcn_wmma_f32_16x16x32_bf16(false, a, false, bB, (short)0, acc1, false, false);
  }

#pragma unroll
  for (int r = 0; r < 8; ++r) {
    int m = row0 + r + 8 * g;
    if (lm < 3) {
      H0[(size_t)m * 3 + lm] = acc0[r];
      H1[(size_t)m * 3 + lm] = acc1[r];
    }
  }
}

// ---------------------------------------------------------------------------
// Edge scatter, dout=128: one wave per directed edge.
// dir 0: H0[src] += H1[dst] ; dir 1: H0[dst] += H1[src]
// ---------------------------------------------------------------------------
__global__ __launch_bounds__(256)
void scatter_add128(const int* __restrict__ edges, const float* __restrict__ Hs,
                    float* __restrict__ Hd, int nDir) {
  int wid = blockIdx.x * 8 + (threadIdx.x >> 5);
  if (wid >= nDir) return;
  int e = wid >> 1, d = wid & 1;
  int a = edges[2 * e], b = edges[2 * e + 1];
  int tgt = d ? b : a;
  int srn = d ? a : b;
  int lane = threadIdx.x & 31;
  const float4 v = *(const float4*)(Hs + (size_t)srn * F + lane * 4);
  float* o = Hd + (size_t)tgt * F + lane * 4;
  atomicAdd(o + 0, v.x);
  atomicAdd(o + 1, v.y);
  atomicAdd(o + 2, v.z);
  atomicAdd(o + 3, v.w);
}

// Edge scatter, dout=3: one thread per directed edge.
__global__ __launch_bounds__(256)
void scatter_add3(const int* __restrict__ edges, const float* __restrict__ Hs,
                  float* __restrict__ Hd, int nDir) {
  int t = blockIdx.x * 256 + threadIdx.x;
  if (t >= nDir) return;
  int e = t >> 1, d = t & 1;
  int a = edges[2 * e], b = edges[2 * e + 1];
  int tgt = d ? b : a;
  int srn = d ? a : b;
  const float* s = Hs + (size_t)srn * 3;
  float* o = Hd + (size_t)tgt * 3;
  atomicAdd(o + 0, s[0]);
  atomicAdd(o + 1, s[1]);
  atomicAdd(o + 2, s[2]);
}

// ---------------------------------------------------------------------------
// Launch: features, edges, then (W0_i, b0_i, W1_i, b1_i) for i = 0..3
// ---------------------------------------------------------------------------
extern "C" void kernel_launch(void* const* d_in, const int* in_sizes, int n_in,
                              void* d_out, int out_size, void* d_ws, size_t ws_size,
                              hipStream_t stream) {
  const float* feat  = (const float*)d_in[0];
  const int*   edges = (const int*)d_in[1];
  const float* W0[4];
  const float* b0[4];
  const float* W1[4];
  const float* b1[4];
  for (int i = 0; i < 4; ++i) {
    W0[i] = (const float*)d_in[2 + 4 * i];
    b0[i] = (const float*)d_in[3 + 4 * i];
    W1[i] = (const float*)d_in[4 + 4 * i];
    b1[i] = (const float*)d_in[5 + 4 * i];
  }
  const int N = in_sizes[0] / F;   // 320000
  const int E = in_sizes[1] / 2;   // 960000
  const size_t S = (size_t)N * F;  // floats per activation slab

  float* ws0 = (float*)d_ws;
  float* ws1 = ws0 + S;
  float* ws2 = ws1 + S;
  __bf16* wt = (__bf16*)(ws2 + S);
  float* out = (float*)d_out;

  const int nDir = 2 * E;
  const int gM16  = N / 16;   // 20000
  const int gM128 = N / 128;  // 2500
  const int gScat = (nDir + 7) / 8;
  const int prepElems = 6 * NSQ + 2 * 16 * F;

  prep_weights<<<(prepElems + 255) / 256, 256, 0, stream>>>(
      W0[0], W1[0], W0[1], W1[1], W0[2], W1[2], W0[3], W1[3], wt);

  // layer 0: A = features (no relu)
  gemm_dual128<<<gM16, 256, 0, stream>>>(feat, wt + 0 * NSQ, b0[0], wt + 1 * NSQ, b1[0],
                                         ws0, ws1, 0);
  scatter_add128<<<gScat, 256, 0, stream>>>(edges, ws1, ws0, nDir);

  // layer 1: A = relu(ws0)
  gemm_dual128<<<gM16, 256, 0, stream>>>(ws0, wt + 2 * NSQ, b0[1], wt + 3 * NSQ, b1[1],
                                         ws1, ws2, 1);
  scatter_add128<<<gScat, 256, 0, stream>>>(edges, ws2, ws1, nDir);

  // layer 2: A = relu(ws1)
  gemm_dual128<<<gM16, 256, 0, stream>>>(ws1, wt + 4 * NSQ, b0[2], wt + 5 * NSQ, b1[2],
                                         ws2, ws0, 1);
  scatter_add128<<<gScat, 256, 0, stream>>>(edges, ws0, ws2, nDir);

  // layer 3: A = relu(ws2) + features, dout = 3 -> d_out (+ H1 in ws1)
  gemm_dual3<<<gM128, 256, 0, stream>>>(ws2, feat,
                                        wt + 6 * NSQ, b0[3],
                                        wt + 6 * NSQ + 16 * F, b1[3],
                                        out, ws1);
  scatter_add3<<<(nDir + 255) / 256, 256, 0, stream>>>(edges, ws1, out, nDir);
}
